// MultiHeadAttention_73237782331757
// MI455X (gfx1250) — compile-verified, matrix operand order checked by reading, not run on hardware
//
#include <hip/hip_runtime.h>
#include <hip/hip_bf16.h>

// ---------------------------------------------------------------------------
// MHA forward for gfx1250 (MI455X): bf16 WMMA GEMMs + fused softmax,
// async-to-LDS (ASYNCcnt) double-buffered staging in the projection GEMMs.
// B=2, S=2048, D_MODEL=1024, H=16, DK=64.
// ---------------------------------------------------------------------------

#define S_LEN   2048
#define DMODEL  1024
#define NHEAD   16
#define DK      64
#define NROWS   4096          // B * S
#define BH      32            // B * H

typedef __attribute__((ext_vector_type(16))) __bf16 v16bf;
typedef __attribute__((ext_vector_type(8)))  __bf16 v8bf;
typedef __attribute__((ext_vector_type(8)))  float  v8f;

// Build a 16-element bf16 WMMA operand (16-bit A/B^T layout) for this lane
// from a bf16 row segment. Lanes 0-15 take K {0..7,16..23}; 16-31 take
// K {8..15,24..31} (per CDNA5 ISA 7.12.2 16-bit A-matrix table).
__device__ inline v16bf load_op16(const __bf16* p, int lane) {
    int hi = (lane >> 4) & 1;
    v8bf x = *(const v8bf*)(p + hi * 8);
    v8bf y = *(const v8bf*)(p + 16 + hi * 8);
    return __builtin_shufflevector(x, y, 0,1,2,3,4,5,6,7,8,9,10,11,12,13,14,15);
}

// Same, but source row is fp32 (packed-convert to bf16 on the fly).
__device__ inline v16bf load_op16_f32(const float* p, int lane) {
    int hi = (lane >> 4) & 1;
    v8f x = *(const v8f*)(p + hi * 8);
    v8f y = *(const v8f*)(p + 16 + hi * 8);
    v8bf lo = __builtin_convertvector(x, v8bf);
    v8bf hb = __builtin_convertvector(y, v8bf);
    return __builtin_shufflevector(lo, hb, 0,1,2,3,4,5,6,7,8,9,10,11,12,13,14,15);
}

__device__ inline v8f wmma_bf16(v16bf a, v16bf b, v8f c) {
    return __builtin_amdgcn_wmma_f32_16x16x32_bf16(
        /*neg_a=*/false, a, /*neg_b=*/false, b,
        /*c_mod=*/(short)0, c, /*reuse_a=*/false, /*reuse_b=*/false);
}

// ---------------------------------------------------------------------------
// Projection GEMM: Y[m,n] = sum_k X[m,k] * W[n,k] + bias[n]
// X: [4096,1024] fp32, W: [1024,1024] fp32.
// Tiles staged to LDS with GLOBAL_LOAD_ASYNC_TO_LDS_B128, double buffered.
// MODE 0: fp32 out [4096,1024]
// MODE 1: bf16 out, head layout [B,H,S,DK]   (q, k)
// MODE 2: bf16 out, transposed  [B,H,DK,S]   (v^T, feeds ctx GEMM B operand)
// Grid: (1024/128, 4096/128) = (8, 32); 256 threads (8 waves).
// ---------------------------------------------------------------------------
template <int MODE>
__global__ __launch_bounds__(256)
void proj_kernel(const float* __restrict__ X, const float* __restrict__ W,
                 const float* __restrict__ bias, void* __restrict__ out) {
    __shared__ float sA[2][128][40];   // X tile,  fp32, 32 K + pad (40 KB)
    __shared__ float sB[2][128][40];   // W rows, fp32, == B^T tile (40 KB)

    const int t    = threadIdx.x;
    const int wid  = t >> 5;
    const int lane = t & 31;
    const int ln   = lane & 15;
    const int wm   = wid >> 2;            // 0..1  -> 64 rows each
    const int wn   = wid & 3;             // 0..3  -> 32 cols each
    const int row0 = blockIdx.y * 128;
    const int col0 = blockIdx.x * 128;

    v8f acc[4][2];
    for (int i = 0; i < 4; ++i)
        for (int j = 0; j < 2; ++j)
            acc[i][j] = (v8f){0.f,0.f,0.f,0.f,0.f,0.f,0.f,0.f};

    // Issue 8 async global->LDS b128 copies per thread for one 128x32 stage.
    auto stage = [&](int buf, int k0) {
        for (int i = 0; i < 4; ++i) {
            int idx = t + i * 256;                    // float4 index 0..1023
            int r = idx >> 3, c4 = (idx & 7) * 4;     // row, col (floats)
            const float* ga = X + (size_t)(row0 + r) * DMODEL + k0 + c4;
            unsigned la = (unsigned)(uintptr_t)&sA[buf][r][c4];
            asm volatile("global_load_async_to_lds_b128 %0, %1, off"
                         :: "v"(la), "v"((unsigned long long)(uintptr_t)ga)
                         : "memory");
            const float* gb = W + (size_t)(col0 + r) * DMODEL + k0 + c4;
            unsigned lb = (unsigned)(uintptr_t)&sB[buf][r][c4];
            asm volatile("global_load_async_to_lds_b128 %0, %1, off"
                         :: "v"(lb), "v"((unsigned long long)(uintptr_t)gb)
                         : "memory");
        }
    };

    stage(0, 0);
    for (int kt = 0; kt < DMODEL / 32; ++kt) {
        const int cur = kt & 1;
        const bool more = (kt + 1) < DMODEL / 32;
        if (more) stage(cur ^ 1, (kt + 1) * 32);      // prefetch next tile
        // Async loads complete in order: <=8 outstanding leaves only the
        // prefetch in flight, current tile is resident in LDS.
        if (more) asm volatile("s_wait_asynccnt 0x8" ::: "memory");
        else      asm volatile("s_wait_asynccnt 0x0" ::: "memory");
        __syncthreads();

        v16bf a[4], b[2];
        for (int tm = 0; tm < 4; ++tm)
            a[tm] = load_op16_f32(&sA[cur][wm * 64 + tm * 16 + ln][0], lane);
        for (int tn = 0; tn < 2; ++tn)
            b[tn] = load_op16_f32(&sB[cur][wn * 32 + tn * 16 + ln][0], lane);

        for (int tm = 0; tm < 4; ++tm)
            for (int tn = 0; tn < 2; ++tn)
                acc[tm][tn] = wmma_bf16(a[tm], b[tn], acc[tm][tn]);
        __syncthreads();   // reads done before next stage overwrites buffer
    }

    // ---- epilogue: bias + layout-specific store ----
    const int mhalf = (lane >> 4) << 3;       // +8 rows for upper half-wave
    for (int tm = 0; tm < 4; ++tm) {
        for (int tn = 0; tn < 2; ++tn) {
            int nn = col0 + wn * 32 + tn * 16 + ln;
            float bv = bias[nn];
            for (int j = 0; j < 8; ++j) {
                int mm = row0 + wm * 64 + tm * 16 + j + mhalf;
                float v = acc[tm][tn][j] + bv;
                if (MODE == 0) {
                    ((float*)out)[(size_t)mm * DMODEL + nn] = v;
                } else {
                    int bb = mm >> 11, ss = mm & (S_LEN - 1);
                    int hh = nn >> 6,  dd = nn & (DK - 1);
                    if (MODE == 1)
                        ((__bf16*)out)[(((size_t)(bb * NHEAD + hh) * S_LEN) + ss) * DK + dd] = (__bf16)v;
                    else
                        ((__bf16*)out)[(((size_t)(bb * NHEAD + hh) * DK) + dd) * S_LEN + ss] = (__bf16)v;
                }
            }
        }
    }
}

// ---------------------------------------------------------------------------
// Fused attention: one WG per (b*h, 16-query block).
// scores (16 x 2048) live in LDS end-to-end: WMMA scores -> softmax ->
// single coalesced fp32 write of attn_weights -> WMMA ctx (all 8 waves,
// K-split with LDS partial reduction).
// Grid: (S/16, BH) = (128, 32); 256 threads (8 waves).
// LDS: 16*2048*4 + 16*72*2 + 4*16*16*4 = ~134.5 KB (< 320 KB WGP budget).
// ---------------------------------------------------------------------------
__global__ __launch_bounds__(256)
void attn_kernel(const __bf16* __restrict__ qb,   // [BH, S, DK]
                 const __bf16* __restrict__ kb,   // [BH, S, DK]
                 const __bf16* __restrict__ vt,   // [BH, DK, S]
                 float* __restrict__ attn_out,    // [BH, S, S]
                 float* __restrict__ cbuf) {      // [NROWS, DMODEL] fp32 ctx
    __shared__ float  sS[16][S_LEN];              // 128 KB score rows
    __shared__ __bf16 sQ[16][72];                 // 16 x 64 q tile (+pad)
    __shared__ float  sP[4][16][16];              // ctx partials (K-split)

    const int t    = threadIdx.x;
    const int wid  = t >> 5;
    const int lane = t & 31;
    const int ln   = lane & 15;
    const int bh   = blockIdx.y;
    const int q0   = blockIdx.x * 16;
    const int bb   = bh >> 4;
    const int hh   = bh & 15;

    // ---- stage q tile ----
    for (int i = t; i < 16 * DK; i += 256) {
        int r = i >> 6, d = i & 63;
        sQ[r][d] = qb[((size_t)bh * S_LEN + q0 + r) * DK + d];
    }
    __syncthreads();

    // ---- scores = q @ k^T * (1/sqrt(DK)); each wave owns 256 key columns ----
    const int mhalf = (lane >> 4) << 3;
    for (int ct = 0; ct < 16; ++ct) {
        int cn = wid * 256 + ct * 16;
        v8f c = (v8f){0.f,0.f,0.f,0.f,0.f,0.f,0.f,0.f};
        for (int ks = 0; ks < DK; ks += 32) {
            v16bf a  = load_op16(&sQ[ln][ks], lane);
            v16bf bm = load_op16(kb + ((size_t)bh * S_LEN + cn + ln) * DK + ks, lane);
            c = wmma_bf16(a, bm, c);
        }
        for (int j = 0; j < 8; ++j)
            sS[j + mhalf][cn + ln] = c[j] * 0.125f;   // 1/sqrt(64)
    }
    __syncthreads();

    // ---- softmax per row (2 rows per wave), write attn_weights once ----
    for (int r = wid * 2; r < wid * 2 + 2; ++r) {
        float mx = -3.402823466e38f;
        for (int cidx = lane; cidx < S_LEN; cidx += 32)
            mx = fmaxf(mx, sS[r][cidx]);
        for (int off = 16; off; off >>= 1)
            mx = fmaxf(mx, __shfl_xor(mx, off, 32));
        float sum = 0.f;
        for (int cidx = lane; cidx < S_LEN; cidx += 32) {
            float e = __expf(sS[r][cidx] - mx);
            sS[r][cidx] = e;
            sum += e;
        }
        for (int off = 16; off; off >>= 1)
            sum += __shfl_xor(sum, off, 32);
        float inv = 1.0f / sum;
        float* orow = attn_out + ((size_t)bh * S_LEN + q0 + r) * S_LEN;
        for (int cidx = lane; cidx < S_LEN; cidx += 32) {
            float p = sS[r][cidx] * inv;
            sS[r][cidx] = p;
            orow[cidx]  = p;
        }
    }
    __syncthreads();

    // ---- ctx = attn @ v : 16x64 output, 4 tiles x 2 K-halves (8 waves) ----
    {
        const int tile = wid & 3;          // which 16-col d tile
        const int kh   = wid >> 2;         // K half: 0 or 1
        const int dn   = tile * 16;
        v8f c = (v8f){0.f,0.f,0.f,0.f,0.f,0.f,0.f,0.f};
        const int hi = (lane >> 4) & 1;
        for (int k0 = kh * (S_LEN / 2); k0 < (kh + 1) * (S_LEN / 2); k0 += 32) {
            v8f x0 = *(const v8f*)&sS[ln][k0 + hi * 8];
            v8f x1 = *(const v8f*)&sS[ln][k0 + 16 + hi * 8];
            v8bf lo = __builtin_convertvector(x0, v8bf);
            v8bf hb = __builtin_convertvector(x1, v8bf);
            v16bf a = __builtin_shufflevector(lo, hb,
                          0,1,2,3,4,5,6,7,8,9,10,11,12,13,14,15);
            v16bf bm = load_op16(vt + ((size_t)bh * DK + dn + ln) * S_LEN + k0, lane);
            c = wmma_bf16(a, bm, c);
        }
        if (kh == 1)
            for (int j = 0; j < 8; ++j)
                sP[tile][j + mhalf][ln] = c[j];
        __syncthreads();
        if (kh == 0) {
            for (int j = 0; j < 8; ++j) {
                int m = q0 + j + mhalf;
                float v = c[j] + sP[tile][j + mhalf][ln];
                cbuf[((size_t)bb * S_LEN + m) * DMODEL + hh * DK + dn + ln] = v;
            }
        }
    }
}

// ---------------------------------------------------------------------------
extern "C" void kernel_launch(void* const* d_in, const int* in_sizes, int n_in,
                              void* d_out, int out_size, void* d_ws, size_t ws_size,
                              hipStream_t stream) {
    const float* Q  = (const float*)d_in[0];
    const float* K  = (const float*)d_in[1];
    const float* V  = (const float*)d_in[2];
    const float* Wq = (const float*)d_in[3];
    const float* bq = (const float*)d_in[4];
    const float* Wk = (const float*)d_in[5];
    const float* bk = (const float*)d_in[6];
    const float* Wv = (const float*)d_in[7];
    const float* bv = (const float*)d_in[8];
    const float* Wo = (const float*)d_in[9];
    const float* bo = (const float*)d_in[10];

    const size_t PROJ_ELEMS = (size_t)BH * S_LEN * DK;   // 4,194,304
    __bf16* qb = (__bf16*)d_ws;
    __bf16* kb = qb + PROJ_ELEMS;
    __bf16* vt = kb + PROJ_ELEMS;
    float* cbuf = (float*)(vt + PROJ_ELEMS);             // [4096,1024] fp32

    float* out  = (float*)d_out;                         // [2,2048,1024]
    float* attn = out + (size_t)NROWS * DMODEL;          // [2,16,2048,2048]

    dim3 pg(DMODEL / 128, NROWS / 128);                  // (8, 32)
    dim3 pb(256);
    hipLaunchKernelGGL((proj_kernel<1>), pg, pb, 0, stream, Q, Wq, bq, (void*)qb);
    hipLaunchKernelGGL((proj_kernel<1>), pg, pb, 0, stream, K, Wk, bk, (void*)kb);
    hipLaunchKernelGGL((proj_kernel<2>), pg, pb, 0, stream, V, Wv, bv, (void*)vt);

    dim3 ag(S_LEN / 16, BH);                             // (128, 32)
    hipLaunchKernelGGL(attn_kernel, ag, pb, 0, stream, qb, kb, vt, attn, cbuf);

    hipLaunchKernelGGL((proj_kernel<0>), pg, pb, 0, stream, cbuf, Wo, bo, (void*)out);
}
